// dhs_vig_36575941492918
// MI455X (gfx1250) — compile-verified, matrix operand order checked by reading, not run on hardware
//
#include <hip/hip_runtime.h>
#include <hip/hip_bf16.h>
#include <math.h>

// ----------------------------------------------------------------------------
// ViG (Vision GNN) forward for MI455X / gfx1250.
// All convs (1x1 = pure GEMM, 3x3 = implicit GEMM) run as bf16 WMMA
// (v_wmma_f32_16x16x32_bf16, fp32 accumulate). Weights are pre-packed to
// K-padded bf16 in workspace so the GEMM hot loop has no guards/conversions
// on the A side. Block tile 64(M) x 128(N), 8 waves; each wave computes a
// 32x32 sub-tile: 2 A-frags x 2 B-frags = 4 WMMAs per K-step (2 ds_load_b128
// per WMMA). Graph construction (distances / top-k / dilated max-aggregate)
// is a wave-per-pixel VALU kernel with LDS scores + shuffle argmin reduction.
// ----------------------------------------------------------------------------

typedef __bf16 bf16_t;
typedef __attribute__((ext_vector_type(16))) __bf16 v16bf;
typedef __attribute__((ext_vector_type(8)))  __bf16 v8bf;
typedef __attribute__((ext_vector_type(8)))  float  v8f;

static __device__ __forceinline__ float gelu_tanh(float v) {
  const float c = 0.7978845608028654f;
  float t = tanhf(c * (v + 0.044715f * v * v * v));
  return 0.5f * v * (1.0f + t);
}

// ----------------------------------------------------------------------------
// Pack f32 weights [Cout][K] -> bf16 [Cout][Kp], zero-padded to Kp (mult. 32).
// ----------------------------------------------------------------------------
__global__ void pack_w_k(const float* __restrict__ w, bf16_t* __restrict__ wp,
                         int K, int Kp, int total) {
  int i = blockIdx.x * 256 + threadIdx.x;
  if (i >= total) return;
  int k = i % Kp, m = i / Kp;
  float v = (k < K) ? w[(size_t)m * K + k] : 0.0f;
  wp[i] = (bf16_t)v;
}

// ----------------------------------------------------------------------------
// Fused conv(1x1 or 3x3) + bias + BN + optional GELU + optional residual.
// Implicit GEMM: M = Cout, N = OH*OW (per image), K = Cin*ks*ks (Kp padded).
// Block = 256 threads (8 waves); tile 64(M) x 128(N); K-step 32.
// LDS rows have 40-elem bf16 pitch so each fragment is two 16B ds_load_b128
// (16-bit A/B VGPR layout: lane's k-pairs live at [koff,koff+8) u [koff+16,+24)).
// ----------------------------------------------------------------------------
#define BM 64
#define BN 128
#define BK 32
#define BKP 40   // LDS row pitch in bf16 elems (80B: preserves 16B alignment)

__launch_bounds__(256)
__global__ void wmma_conv_bn_k(const float* __restrict__ in,
                               const bf16_t* __restrict__ wp,
                               const float* __restrict__ bias,
                               const float* __restrict__ gg,
                               const float* __restrict__ be,
                               const float* __restrict__ res,
                               float* __restrict__ out,
                               int Cin, int Cout, int H, int W,
                               int OH, int OW, int ks, int stride, int pad,
                               int Kp, int gelu) {
  __shared__ bf16_t As[BM][BKP];
  __shared__ bf16_t Bs[BN][BKP];

  const int Npix = OH * OW;
  const int K    = Cin * ks * ks;
  const int b    = blockIdx.z;
  const int m0   = blockIdx.y * BM;
  const int n0   = blockIdx.x * BN;
  const int tid  = threadIdx.x;
  const int lane = tid & 31;
  const int wv   = tid >> 5;          // wave id 0..7
  const int mo   = (wv & 1) * 32;     // wave's M sub-tile base (32 rows)
  const int nb   = (wv >> 1) * 32;    // wave's N sub-tile base (32 cols)

  const float* inb = in + (size_t)b * Cin * H * W;

  // A loader mapping: 64x32 bf16 tile, 8 elems (16B) per thread
  const int arow = tid >> 2;          // 0..63
  const int acol = (tid & 3) * 8;
  // B loader mapping: one k per thread, 16 pixels
  const int kl = tid >> 3;            // 0..31
  const int ng = (tid & 7) * 16;      // pixel group base 0..112

  v8f acc00 = {}, acc01 = {}, acc10 = {}, acc11 = {};

  for (int kk = 0; kk < Kp; kk += BK) {
    // ---- stage A tile from pre-packed bf16 weights (no k guard) ----
    {
      int m = m0 + arow;
      v8bf av = {};
      if (m < Cout) av = *(const v8bf*)(wp + (size_t)m * Kp + kk + acol);
      *(v8bf*)&As[arow][acol] = av;
      __builtin_prefetch(wp + (size_t)m * Kp + kk + BK, 0, 1);
    }
    // ---- stage B tile: [k][16 pixels] per thread ----
    {
      int k = kk + kl;
      bool kin = (k < K);
      if (ks == 1 && kin && (n0 + BN <= Npix) && ((Npix & 3) == 0)) {
        // fastest path: 16B-aligned vector loads (Npix mult of 4)
        const float4* src = (const float4*)(inb + (size_t)k * Npix + n0 + ng);
        #pragma unroll
        for (int j4 = 0; j4 < 4; ++j4) {
          float4 v = src[j4];
          Bs[ng + j4 * 4 + 0][kl] = (bf16_t)v.x;
          Bs[ng + j4 * 4 + 1][kl] = (bf16_t)v.y;
          Bs[ng + j4 * 4 + 2][kl] = (bf16_t)v.z;
          Bs[ng + j4 * 4 + 3][kl] = (bf16_t)v.w;
        }
      } else if (ks == 1) {
        #pragma unroll
        for (int j = 0; j < 16; ++j) {
          int npix = n0 + ng + j;
          float v = (kin && npix < Npix) ? inb[(size_t)k * Npix + npix] : 0.0f;
          Bs[ng + j][kl] = (bf16_t)v;
        }
      } else {  // 3x3 im2col: decode k once per thread
        int ci = 0, kh = 0, kw = 0;
        if (kin) { ci = k / 9; int rem = k - ci * 9; kh = rem / 3; kw = rem - kh * 3; }
        #pragma unroll
        for (int j = 0; j < 16; ++j) {
          int npix = n0 + ng + j;
          float v = 0.0f;
          if (kin && npix < Npix) {
            int oh = npix / OW, ow = npix - oh * OW;
            int ih = oh * stride - pad + kh;
            int iw = ow * stride - pad + kw;
            if ((unsigned)ih < (unsigned)H && (unsigned)iw < (unsigned)W)
              v = inb[((size_t)ci * H + ih) * W + iw];
          }
          Bs[ng + j][kl] = (bf16_t)v;
        }
      }
    }
    __syncthreads();

    // ---- fragments per documented 16-bit A/B VGPR layout ----
    {
      const int half = lane >> 4;
      const int l16  = lane & 15;
      const int koff = half * 8;
      const bf16_t* ar0 = &As[mo + l16][0];
      const bf16_t* ar1 = &As[mo + 16 + l16][0];
      const bf16_t* br0 = &Bs[nb + l16][0];
      const bf16_t* br1 = &Bs[nb + 16 + l16][0];
      v8bf t0, t1;
      t0 = *(const v8bf*)(ar0 + koff); t1 = *(const v8bf*)(ar0 + koff + 16);
      v16bf a0 = __builtin_shufflevector(t0, t1, 0,1,2,3,4,5,6,7,8,9,10,11,12,13,14,15);
      t0 = *(const v8bf*)(ar1 + koff); t1 = *(const v8bf*)(ar1 + koff + 16);
      v16bf a1 = __builtin_shufflevector(t0, t1, 0,1,2,3,4,5,6,7,8,9,10,11,12,13,14,15);
      t0 = *(const v8bf*)(br0 + koff); t1 = *(const v8bf*)(br0 + koff + 16);
      v16bf b0 = __builtin_shufflevector(t0, t1, 0,1,2,3,4,5,6,7,8,9,10,11,12,13,14,15);
      t0 = *(const v8bf*)(br1 + koff); t1 = *(const v8bf*)(br1 + koff + 16);
      v16bf b1 = __builtin_shufflevector(t0, t1, 0,1,2,3,4,5,6,7,8,9,10,11,12,13,14,15);
      acc00 = __builtin_amdgcn_wmma_f32_16x16x32_bf16(false, a0, false, b0,
                                                      (short)0, acc00, false, false);
      acc01 = __builtin_amdgcn_wmma_f32_16x16x32_bf16(false, a0, false, b1,
                                                      (short)0, acc01, false, false);
      acc10 = __builtin_amdgcn_wmma_f32_16x16x32_bf16(false, a1, false, b0,
                                                      (short)0, acc10, false, false);
      acc11 = __builtin_amdgcn_wmma_f32_16x16x32_bf16(false, a1, false, b1,
                                                      (short)0, acc11, false, false);
    }
    __syncthreads();
  }

  // ---- epilogue: C/D layout (VGPR r: M = r + 8*half, N = lane&15) ----
  {
    const int half = lane >> 4;
    const int l16  = lane & 15;
    #pragma unroll
    for (int am = 0; am < 2; ++am) {
      #pragma unroll
      for (int sub = 0; sub < 2; ++sub) {
        int npix = n0 + nb + sub * 16 + l16;
        if (npix < Npix) {
          const v8f& a = am ? (sub ? acc11 : acc10) : (sub ? acc01 : acc00);
          #pragma unroll
          for (int r = 0; r < 8; ++r) {
            int co = m0 + mo + am * 16 + r + (half ? 8 : 0);
            if (co < Cout) {
              float v = (a[r] + bias[co]) * gg[co] + be[co];
              if (gelu) v = gelu_tanh(v);
              size_t o = ((size_t)b * Cout + co) * Npix + npix;
              if (res) v += res[o];
              out[o] = v;
            }
          }
        }
      }
    }
  }
}

// ----------------------------------------------------------------------------
// x += pos_embed (broadcast over batch)
// ----------------------------------------------------------------------------
__global__ void addpos_k(float* __restrict__ x, const float* __restrict__ pos,
                         int per, int total) {
  int i = blockIdx.x * 256 + threadIdx.x;
  if (i < total) x[i] += pos[i % per];
}

// ----------------------------------------------------------------------------
// r x r average pooling (NCHW), out [B,C,Hm,Wm]
// ----------------------------------------------------------------------------
__global__ void avgpool_k(const float* __restrict__ in, float* __restrict__ out,
                          int C, int H, int W, int r, int total) {
  int i = blockIdx.x * 256 + threadIdx.x;
  if (i >= total) return;
  int Hm = H / r, Wm = W / r, M = Hm * Wm;
  int m = i % M; int t = i / M; int c = t % C; int b = t / C;
  int mh = m / Wm, mw = m - mh * Wm;
  const float* p = in + (((size_t)b * C + c) * H + mh * r) * W + mw * r;
  float s = 0.0f;
  for (int a = 0; a < r; ++a)
    for (int q = 0; q < r; ++q) s += p[a * W + q];
  out[((size_t)b * C + c) * M + m] = s / (float)(r * r);
}

// ----------------------------------------------------------------------------
// yn[b,m] = sum_c y[b,c,m]^2
// ----------------------------------------------------------------------------
__global__ void sqnorm_k(const float* __restrict__ y, float* __restrict__ yn,
                         int C, int M, int total) {
  int i = blockIdx.x * 256 + threadIdx.x;
  if (i >= total) return;
  int m = i % M; int b = i / M;
  const float* p = y + (size_t)b * C * M + m;
  float s = 0.0f;
  for (int c = 0; c < C; ++c) { float v = p[(size_t)c * M]; s = fmaf(v, v, s); }
  yn[i] = s;
}

// ----------------------------------------------------------------------------
// Graph feature: per pixel n, score[m] = -2 x_n . y_m + |y_m|^2 + rel[n,m]
// (|x_n|^2 dropped: constant w.r.t. ranking). Select k*dil smallest (ties ->
// lowest index, matching lax.top_k), keep every dil-th, aggregate
// m[c] = max_j y[c, sel_j] - x[c, n].  Output interleaved: ch 2c = x, 2c+1 = m
// (matches stack(...,-1).reshape(B,N,2C)).  One wave per (n, b).
// ----------------------------------------------------------------------------
__launch_bounds__(32)
__global__ void graph_feat_k(const float* __restrict__ xt,
                             const float* __restrict__ yt,
                             const float* __restrict__ ynorm,
                             const float* __restrict__ rel,
                             float* __restrict__ out,
                             int C, int N, int M, int k, int dil) {
  __shared__ float sc[256];    // M <= 196
  __shared__ float xsh[704];   // C <= 640
  __shared__ int   sel[64];    // k*dil <= 27

  const int n    = blockIdx.x;
  const int b    = blockIdx.y;
  const int lane = threadIdx.x;
  const float* xb = xt + (size_t)b * C * N;
  const float* yb = yt + (size_t)b * C * M;

  for (int c = lane; c < C; c += 32) xsh[c] = xb[(size_t)c * N + n];
  __syncthreads();

  for (int m = lane; m < M; m += 32) {
    float dot = 0.0f;
    for (int c = 0; c < C; ++c) dot = fmaf(xsh[c], yb[(size_t)c * M + m], dot);
    sc[m] = ynorm[b * M + m] - 2.0f * dot + rel[(size_t)n * M + m];
  }
  __syncthreads();

  const int kd = k * dil;
  for (int it = 0; it < kd; ++it) {
    float bv = 3.4e38f; int bi = 0x7fffffff;
    for (int m = lane; m < M; m += 32) {
      float v = sc[m];
      if (v < bv) { bv = v; bi = m; }
    }
    #pragma unroll
    for (int off = 16; off > 0; off >>= 1) {
      float ov = __shfl_xor(bv, off, 32);
      int   oi = __shfl_xor(bi, off, 32);
      if (ov < bv || (ov == bv && oi < bi)) { bv = ov; bi = oi; }
    }
    if (lane == 0) {
      if (bi == 0x7fffffff) bi = 0;
      sel[it] = bi;
      sc[bi] = 3.4e38f;
    }
    __syncthreads();
  }

  for (int c = lane; c < C; c += 32) {
    float xv = xsh[c];
    float mx = -3.4e38f;
    for (int j = 0; j < k; ++j) {
      int m = sel[j * dil];
      mx = fmaxf(mx, yb[(size_t)c * M + m]);
    }
    size_t base = (size_t)b * 2 * C * N;
    out[base + (size_t)(2 * c)     * N + n] = xv;
    out[base + (size_t)(2 * c + 1) * N + n] = mx - xv;
  }
}

// ----------------------------------------------------------------------------
// Host orchestration
// ----------------------------------------------------------------------------
extern "C" void kernel_launch(void* const* d_in, const int* in_sizes, int n_in,
                              void* d_out, int out_size, void* d_ws, size_t ws_size,
                              hipStream_t stream) {
  (void)in_sizes; (void)n_in; (void)out_size; (void)ws_size;
  const int B = 16;
  const int chan[4]  = {80, 160, 400, 640};
  const int ratio[4] = {4, 2, 1, 1};

  // --- input index map (recursive insertion-order flatten of setup_inputs) ---
  // 0: x ; 1..12: stem c1,c2,c3 (w,b,g,be each) ; 13: pos_embed ;
  // 14..: 15 backbone entries (block=20 arrays, down=4) ; then 12 rel_pos.
  const int isDown[15] = {0,0,1,0,0,1,0,0,0,0,0,0,1,0,0};
  int entryBase[15];
  int base = 14;
  for (int i = 0; i < 15; ++i) { entryBase[i] = base; base += isDown[i] ? 4 : 20; }
  const int relBase = base;  // 266

  const float* X_IN = (const float*)d_in[0];
  float* out = (float*)d_out;

  // --- workspace carve (needs ~330 MB) ---
  const size_t MAXELEM = 16056320;  // 16*320*3136 (largest intermediate)
  float* bufX  = (float*)d_ws;
  float* bufT1 = bufX  + MAXELEM;
  float* bufT2 = bufT1 + MAXELEM;
  float* bufT3 = bufT2 + MAXELEM;
  float* bufY  = bufT3 + MAXELEM;   // pooled y, <= 16*160*196
  float* bufYN = bufY  + 524288;    // |y|^2, <= 16*196
  bf16_t* bufWP = (bf16_t*)(bufYN + 4096);  // packed weights, <= ~31M bf16
  size_t wpCur = 0;

  auto conv = [&](const float* in, float* o, int pw, int Cin, int Cout,
                  int H, int W, int OH, int OW, int ks, int stride, int pad,
                  int gelu, const float* res) {
    int K = Cin * ks * ks;
    int Kp = (K + 31) & ~31;
    bf16_t* wp = bufWP + wpCur;
    wpCur += (size_t)Cout * Kp;
    int ptot = Cout * Kp;
    pack_w_k<<<(ptot + 255) / 256, 256, 0, stream>>>((const float*)d_in[pw], wp, K, Kp, ptot);
    dim3 grid((OH * OW + BN - 1) / BN, (Cout + BM - 1) / BM, B);
    wmma_conv_bn_k<<<grid, 256, 0, stream>>>(
        in, wp, (const float*)d_in[pw + 1],
        (const float*)d_in[pw + 2], (const float*)d_in[pw + 3],
        res, o, Cin, Cout, H, W, OH, OW, ks, stride, pad, Kp, gelu);
  };
  auto conv1 = [&](const float* in, float* o, int pw, int Cin, int Cout,
                   int H, int gelu, const float* res) {
    conv(in, o, pw, Cin, Cout, H, H, H, H, 1, 1, 0, gelu, res);
  };

  // --- stem: 3 -> 40 -> 80 -> 80, gelu on first two ---
  conv(X_IN, bufT1, 1, 3, 40, 224, 224, 112, 112, 3, 2, 1, 1, nullptr);
  conv(bufT1, bufT2, 5, 40, 80, 112, 112, 56, 56, 3, 2, 1, 1, nullptr);
  conv(bufT2, bufX, 9, 80, 80, 56, 56, 56, 56, 3, 1, 1, 0, nullptr);
  {
    int per = 80 * 56 * 56, total = B * per;
    addpos_k<<<(total + 255) / 256, 256, 0, stream>>>(bufX, (const float*)d_in[13], per, total);
  }

  // --- backbone ---
  float* cur = bufX;
  float* aux = bufT1;
  int s = 0, C = 80, H = 56, bi = 0;
  size_t outOff = 0;

  for (int i = 0; i < 15; ++i) {
    int pb = entryBase[i];
    if (isDown[i]) {
      ++s;
      int Cout = chan[s], OHs = H / 2;
      conv(cur, aux, pb, C, Cout, H, H, OHs, OHs, 3, 2, 1, 0, nullptr);
      float* t = cur; cur = aux; aux = t;
      C = Cout; H = OHs;
    } else {
      int N = H * H;
      int r = ratio[s];
      int Hm = H / r, M = Hm * Hm;
      int dil = bi / 4 + 1; if (dil > 5) dil = 5;

      // grapher
      conv1(cur, aux, pb, C, C, H, 0, nullptr);                 // fc1 -> aux
      const float* Yp;
      if (r > 1) {
        int total = B * C * M;
        avgpool_k<<<(total + 255) / 256, 256, 0, stream>>>(aux, bufY, C, H, H, r, total);
        Yp = bufY;
      } else {
        Yp = aux;
      }
      {
        int total = B * M;
        sqnorm_k<<<(total + 255) / 256, 256, 0, stream>>>(Yp, bufYN, C, M, total);
      }
      graph_feat_k<<<dim3(N, B), 32, 0, stream>>>(
          aux, Yp, bufYN, (const float*)d_in[relBase + bi], bufT2, C, N, M, 9, dil);
      conv1(bufT2, bufT3, pb + 4, 2 * C, 2 * C, H, 1, nullptr); // gc (gelu)
      conv1(bufT3, cur, pb + 8, 2 * C, C, H, 0, cur);           // fc2 + shortcut

      // ffn
      conv1(cur, aux, pb + 12, C, 4 * C, H, 1, nullptr);        // fc1 (gelu)
      conv1(aux, cur, pb + 16, 4 * C, C, H, 0, cur);            // fc2 + shortcut
      ++bi;
    }

    if (i == 1 || i == 4 || i == 11) {
      size_t elems = (size_t)B * C * H * H;
      hipMemcpyAsync(out + outOff, cur, elems * sizeof(float),
                     hipMemcpyDeviceToDevice, stream);
      outOff += elems;
    }
  }
  // final output
  {
    size_t elems = (size_t)B * C * H * H;
    hipMemcpyAsync(out + outOff, cur, elems * sizeof(float),
                   hipMemcpyDeviceToDevice, stream);
  }
}
